// MultiSrcJointMultiHeadedAttention_24103356465716
// MI455X (gfx1250) — matrix operand order repaired and checked by reading, not compiled, not run on hardware
//
#include <hip/hip_runtime.h>
#include <math.h>
#include <stdint.h>

typedef unsigned short ushort_t;
typedef __attribute__((ext_vector_type(16))) __bf16 v16bf;
typedef __attribute__((ext_vector_type(8)))  __bf16 v8bf;
typedef __attribute__((ext_vector_type(8)))  float  v8f;

union BF16x16 { v16bf v; v8bf h[2]; ushort_t u[16]; };

#define Bb   4
#define Ss   1024
#define C1n  512
#define C2n  512
#define KTn  2048
#define Dn   1024
#define Hn   16
#define DHn  64
#define SCALE_Q 0.125f

#if __has_builtin(__builtin_amdgcn_global_load_async_to_lds_b128) && __has_builtin(__builtin_amdgcn_s_wait_asynccnt)
#define HAVE_ASYNC 1
#else
#define HAVE_ASYNC 0
#endif

// exact pointee type of the async-load builtin params: int __vector(4)
typedef int v4i_vs __attribute__((vector_size(16)));
typedef __attribute__((address_space(1))) v4i_vs as1_v4i;
typedef __attribute__((address_space(3))) v4i_vs as3_v4i;

__device__ __forceinline__ as1_v4i* to_global(const void* p) {
    return (as1_v4i*)(unsigned long long)(uintptr_t)p;
}
__device__ __forceinline__ as3_v4i* to_lds(const void* p) {
    // generic pointers to LDS carry the LDS offset in the low 32 bits (ISA 10.2)
    return (as3_v4i*)(unsigned int)(uintptr_t)p;
}

__device__ __forceinline__ ushort_t f2bf(float x) {
    unsigned u = __float_as_uint(x);
    unsigned r = u + 0x7FFFu + ((u >> 16) & 1u);   // round-to-nearest-even
    return (ushort_t)(r >> 16);
}

__device__ __forceinline__ v8bf ld8(const ushort_t* p) {
    return *(const v8bf*)p;
}

__device__ __forceinline__ v8f wmma_bf16(v16bf a, v16bf b, v8f c) {
    return __builtin_amdgcn_wmma_f32_16x16x32_bf16(false, a, false, b, (short)0, c, false, false);
}

// ---------------- conversion kernels ----------------
__global__ void cvt_bf16_kernel(const float* __restrict__ in, ushort_t* __restrict__ out, int n) {
    int i = blockIdx.x * 256 + threadIdx.x;
    if (i < n) out[i] = f2bf(in[i]);
}

// W (K=1024, N=1024) f32 row-major  ->  WT (N,K) bf16 row-major
__global__ void cvt_tr_kernel(const float* __restrict__ W, ushort_t* __restrict__ WT) {
    int i = blockIdx.x * 256 + threadIdx.x;         // over 1024*1024
    int k = i >> 10, n = i & 1023;
    WT[(size_t)n * 1024 + k] = f2bf(W[i]);
}

// ---------------- WMMA GEMM: out = A(M,1024) @ W(1024,1024) + b ----------------
// One wave computes a 16x64 tile: A-fragment reused across 4 accumulators.
// mode 0: f32 out[row*1024+col]
// mode 1: bf16 q layout  (B,H,S,DH),  value*scale after bias
// mode 2: bf16 k layout  (B,H,KT,DH) at row base
// mode 3: bf16 vT layout (B,H,DH,KT) at row base
__global__ void gemm16_kernel(const ushort_t* __restrict__ A, const ushort_t* __restrict__ WT,
                              const float* __restrict__ bias,
                              int rowsPerBatch, int base, int mode, float scale,
                              ushort_t* __restrict__ outB, float* __restrict__ outF) {
    const int wave = threadIdx.x >> 5;
    const int lane = threadIdx.x & 31;
    const int lo = lane & 15, hi = lane >> 4;
    const int nb = (blockIdx.x * 4 + wave) * 64;    // 64-col slab, grid.x = 4
    const int mtile = blockIdx.y;
    const int arow = mtile * 16 + lo;

    const ushort_t* Ap = A  + (size_t)arow * 1024;
    const ushort_t* Wp = WT + (size_t)(nb + lo) * 1024;   // j-th column tile at +j*16*1024

    v8f acc0 = {0,0,0,0,0,0,0,0}, acc1 = acc0, acc2 = acc0, acc3 = acc0;
    for (int kk = 0; kk < 1024; kk += 32) {
        BF16x16 a, b0, b1, b2, b3;
        a.h[0] = ld8(Ap + kk + 16 * hi);
        a.h[1] = ld8(Ap + kk + 16 * hi + 8);
        b0.h[0] = ld8(Wp + 0 * 16384 + kk + 16 * hi); b0.h[1] = ld8(Wp + 0 * 16384 + kk + 16 * hi + 8);
        b1.h[0] = ld8(Wp + 1 * 16384 + kk + 16 * hi); b1.h[1] = ld8(Wp + 1 * 16384 + kk + 16 * hi + 8);
        b2.h[0] = ld8(Wp + 2 * 16384 + kk + 16 * hi); b2.h[1] = ld8(Wp + 2 * 16384 + kk + 16 * hi + 8);
        b3.h[0] = ld8(Wp + 3 * 16384 + kk + 16 * hi); b3.h[1] = ld8(Wp + 3 * 16384 + kk + 16 * hi + 8);
        acc0 = wmma_bf16(a.v, b0.v, acc0);
        acc1 = wmma_bf16(a.v, b1.v, acc1);
        acc2 = wmma_bf16(a.v, b2.v, acc2);
        acc3 = wmma_bf16(a.v, b3.v, acc3);
    }

    const int hh = nb >> 6;                          // whole 64-col slab is one head
    #pragma unroll
    for (int j = 0; j < 4; ++j) {
        v8f acc = (j == 0) ? acc0 : (j == 1) ? acc1 : (j == 2) ? acc2 : acc3;
        const int col = nb + j * 16 + lo;
        const int dh = j * 16 + lo;
        const float bv = bias[col];
        #pragma unroll
        for (int r = 0; r < 8; ++r) {
            int orow = mtile * 16 + r + 8 * hi;
            float val = (acc[r] + bv) * scale;
            if (mode == 0) {
                outF[(size_t)orow * 1024 + col] = val;
            } else {
                ushort_t bx = f2bf(val);
                int bi = orow / rowsPerBatch, sl = orow % rowsPerBatch;
                size_t bh = (size_t)bi * Hn + hh;
                if (mode == 1)      outB[(bh * Ss  + sl) * DHn + dh] = bx;
                else if (mode == 2) outB[(bh * KTn + base + sl) * DHn + dh] = bx;
                else                outB[(bh * DHn + dh) * KTn + base + sl] = bx;
            }
        }
    }
}

// ---------------- flash attention ----------------
// Block = 4 waves sharing one (b,h); each wave owns a 16-row q-tile.
// K/V chunks (32 cols) staged once per block into LDS, double-buffered via
// async-to-LDS DMA (ASYNCcnt) when available.
__global__ void attn_kernel(const ushort_t* __restrict__ qbf, const ushort_t* __restrict__ kbf,
                            const ushort_t* __restrict__ vT,
                            const unsigned char* __restrict__ mS,
                            const unsigned char* __restrict__ m0,
                            const unsigned char* __restrict__ m1,
                            const float* __restrict__ ctx_bias_p,
                            float* __restrict__ attn_raw, float* __restrict__ stats,
                            ushort_t* __restrict__ ctxbf) {
    __shared__ __align__(16) ushort_t kst[2][32 * 64];   // [buf][krow][dh]
    __shared__ __align__(16) ushort_t vst[2][64 * 32];   // [buf][dh][krow]
    __shared__ __align__(16) ushort_t pst[4][16 * 32];   // per-wave P staging

    const int wave = threadIdx.x >> 5;
    const int lane = threadIdx.x & 31;
    const int lo = lane & 15, hi = lane >> 4;
    const int t = threadIdx.x;
    const int gw = blockIdx.x * 4 + wave;           // over B*H*(S/16) = 4096
    const int qt = gw & 63;
    const int bh = gw >> 6;                         // block-uniform (4 | 64)
    const int b  = bh >> 4;
    const float cbias = ctx_bias_p[0];

    const ushort_t* kb = kbf + (size_t)bh * KTn * DHn;
    const ushort_t* vb = vT  + (size_t)bh * DHn * KTn;

    // q A-fragments (DH=64 -> two 16x32 fragments), pre-scaled by 1/sqrt(DH)
    const ushort_t* qp = qbf + ((size_t)bh * Ss + qt * 16 + lo) * DHn;
    BF16x16 qa0, qa1;
    qa0.h[0] = ld8(qp + 16 * hi);      qa0.h[1] = ld8(qp + 16 * hi + 8);
    qa1.h[0] = ld8(qp + 32 + 16 * hi); qa1.h[1] = ld8(qp + 32 + 16 * hi + 8);

    float m[8], l[8], lc[8];
    #pragma unroll
    for (int r = 0; r < 8; ++r) { m[r] = -INFINITY; l[r] = 0.f; lc[r] = 0.f; }
    v8f O0 = {0,0,0,0,0,0,0,0}, O1 = O0, O2 = O0, O3 = O0;

    float* arow = attn_raw + ((size_t)bh * Ss + qt * 16) * KTn;

    // cooperative stage of one 32-col chunk: k 4KB + v 4KB, 128 threads x 2 b128 each
    auto stage = [&](int kcx, int bufx) {
        #pragma unroll
        for (int j = 0; j < 2; ++j) {
            const int c = 2 * t + j;                            // 0..255
            const ushort_t* ks = kb + (size_t)kcx * DHn + c * 8;                        // contiguous 4KB
            const ushort_t* vs = vb + (size_t)(c >> 2) * KTn + kcx + (c & 3) * 8;       // 64 rows x 64B
#if HAVE_ASYNC
            __builtin_amdgcn_global_load_async_to_lds_b128(to_global(ks), to_lds(&kst[bufx][c * 8]), 0, 0);
            __builtin_amdgcn_global_load_async_to_lds_b128(to_global(vs), to_lds(&vst[bufx][c * 8]), 0, 0);
#else
            *(v8bf*)&kst[bufx][c * 8] = ld8(ks);
            *(v8bf*)&vst[bufx][c * 8] = ld8(vs);
#endif
        }
    };

#if HAVE_ASYNC
    stage(0, 0);
#endif

    for (int ic = 0; ic < 64; ++ic) {
        const int kc = ic * 32;
        const bool isctx = (kc >= Ss);
#if HAVE_ASYNC
        const int buf = ic & 1;
        __builtin_amdgcn_s_wait_asynccnt(0);        // my chunk-ic loads landed
        __syncthreads();                            // everyone's landed; prior reads done
        if (ic < 63) stage(kc + 32, buf ^ 1);       // overlap DMA of next chunk
#else
        const int buf = 0;
        __syncthreads();                            // prior iteration's reads done
        stage(kc, 0);
        if (ic < 63) {                              // prefetch next chunk -> global_prefetch_b8
            __builtin_prefetch(kb + (size_t)(kc + 32) * DHn + (size_t)t * 32, 0, 0);
            __builtin_prefetch(vb + (size_t)(t >> 1) * KTn + kc + 32 + (t & 1) * 16, 0, 0);
        }
        __syncthreads();
#endif

        // QK^T: two 16-col halves from LDS
        v8f s0 = {0,0,0,0,0,0,0,0}, s1 = s0;
        {
            const ushort_t* kr = &kst[buf][lo * 64];
            BF16x16 k0, k1;
            k0.h[0] = *(const v8bf*)(kr + 16 * hi);      k0.h[1] = *(const v8bf*)(kr + 16 * hi + 8);
            k1.h[0] = *(const v8bf*)(kr + 32 + 16 * hi); k1.h[1] = *(const v8bf*)(kr + 32 + 16 * hi + 8);
            s0 = wmma_bf16(qa0.v, k0.v, s0);
            s0 = wmma_bf16(qa1.v, k1.v, s0);
        }
        {
            const ushort_t* kr = &kst[buf][(16 + lo) * 64];
            BF16x16 k0, k1;
            k0.h[0] = *(const v8bf*)(kr + 16 * hi);      k0.h[1] = *(const v8bf*)(kr + 16 * hi + 8);
            k1.h[0] = *(const v8bf*)(kr + 32 + 16 * hi); k1.h[1] = *(const v8bf*)(kr + 32 + 16 * hi + 8);
            s1 = wmma_bf16(qa0.v, k0.v, s1);
            s1 = wmma_bf16(qa1.v, k1.v, s1);
        }

        // bias + mask for this lane's two columns
        const int col0 = kc + lo, col1 = kc + 16 + lo;
        const float add = isctx ? cbias : 0.f;
        unsigned char mk0, mk1;
        if (col0 < Ss)            mk0 = mS[b * Ss + col0];
        else if (col0 < Ss + C1n) mk0 = m0[b * C1n + col0 - Ss];
        else                      mk0 = m1[b * C2n + col0 - Ss - C1n];
        if (col1 < Ss)            mk1 = mS[b * Ss + col1];
        else if (col1 < Ss + C1n) mk1 = m0[b * C1n + col1 - Ss];
        else                      mk1 = m1[b * C2n + col1 - Ss - C1n];

        float ps0[8], ps1[8];
        #pragma unroll
        for (int r = 0; r < 8; ++r) {
            float a0 = mk0 ? -1e18f : s0[r] + add;
            float a1 = mk1 ? -1e18f : s1[r] + add;
            float mx = fmaxf(a0, a1);
            #pragma unroll
            for (int msk = 1; msk < 16; msk <<= 1) mx = fmaxf(mx, __shfl_xor(mx, msk, 32));
            float mnew = fmaxf(m[r], mx);
            float sc = __expf(m[r] - mnew);
            float p0 = __expf(a0 - mnew);
            float p1 = __expf(a1 - mnew);
            float sum = p0 + p1;
            #pragma unroll
            for (int msk = 1; msk < 16; msk <<= 1) sum += __shfl_xor(sum, msk, 32);
            l[r] = l[r] * sc + sum;
            lc[r] = isctx ? (lc[r] * sc + sum) : (lc[r] * sc);
            m[r] = mnew;
            O0[r] *= sc; O1[r] *= sc; O2[r] *= sc; O3[r] *= sc;
            ps0[r] = p0; ps1[r] = p1;
            // raw masked scores -> attn output region (normalized by later pass)
            arow[(size_t)(r + 8 * hi) * KTn + col0] = a0;
            arow[(size_t)(r + 8 * hi) * KTn + col1] = a1;
        }

        // stage P (16x32) through LDS: C-fragment layout -> A-fragment layout
        #pragma unroll
        for (int r = 0; r < 8; ++r) {
            pst[wave][(r + 8 * hi) * 32 + lo]      = f2bf(ps0[r]);
            pst[wave][(r + 8 * hi) * 32 + 16 + lo] = f2bf(ps1[r]);
        }
        __syncthreads();
        BF16x16 pa;
        pa.h[0] = *(const v8bf*)&pst[wave][lo * 32 + 16 * hi];
        pa.h[1] = *(const v8bf*)&pst[wave][lo * 32 + 16 * hi + 8];

        // PV from LDS (v transposed: [dh][krow])
        #pragma unroll
        for (int f = 0; f < 4; ++f) {
            const ushort_t* vr = &vst[buf][(f * 16 + lo) * 32];
            BF16x16 vf;
            vf.h[0] = *(const v8bf*)(vr + 16 * hi);
            vf.h[1] = *(const v8bf*)(vr + 16 * hi + 8);
            v8f o = (f == 0) ? O0 : (f == 1) ? O1 : (f == 2) ? O2 : O3;
            o = wmma_bf16(pa.v, vf.v, o);
            if (f == 0) O0 = o; else if (f == 1) O1 = o; else if (f == 2) O2 = o; else O3 = o;
        }
    }

    // epilogue: ctx = O / l  -> ctxbf (B,S,D) bf16
    const int h = bh & 15;
    #pragma unroll
    for (int r = 0; r < 8; ++r) {
        float inv = 1.f / l[r];
        int qrow = qt * 16 + r + 8 * hi;
        size_t obase = ((size_t)b * Ss + qrow) * Dn + h * DHn + lo;
        ctxbf[obase + 0 * 16] = f2bf(O0[r] * inv);
        ctxbf[obase + 1 * 16] = f2bf(O1[r] * inv);
        ctxbf[obase + 2 * 16] = f2bf(O2[r] * inv);
        ctxbf[obase + 3 * 16] = f2bf(O3[r] * inv);
    }
    if (lo == 0) {
        #pragma unroll
        for (int r = 0; r < 8; ++r) {
            size_t row = (size_t)bh * Ss + qt * 16 + r + 8 * hi;
            stats[row * 4 + 0] = m[r];
            stats[row * 4 + 1] = l[r];
            stats[row * 4 + 2] = lc[r];
        }
    }
}

// ---------------- normalize raw scores in place; emit top_attn for head 0 ----------------
__global__ void norm_attn_kernel(float* __restrict__ attn, const float* __restrict__ stats,
                                 float* __restrict__ top) {
    size_t i = (size_t)blockIdx.x * 256 + threadIdx.x;   // over B*H*S*KT = 134217728
    int col = (int)(i & (KTn - 1));
    size_t row = i >> 11;
    float s  = attn[i];
    float mm = stats[row * 4 + 0];
    float ll = stats[row * 4 + 1];
    float lx = stats[row * 4 + 2];
    float tv = __expf(s - mm);
    attn[i] = tv / ll;
    if (col >= Ss && (((row >> 10) & 15) == 0)) {        // head 0 rows
        size_t b = row >> 14;
        size_t q = row & 1023;
        top[(b * Ss + q) * (C1n + C2n) + (col - Ss)] = tv / (lx + 1e-20f * ll);
    }
}

// ---------------- launcher ----------------
extern "C" void kernel_launch(void* const* d_in, const int* in_sizes, int n_in,
                              void* d_out, int out_size, void* d_ws, size_t ws_size,
                              hipStream_t stream) {
    (void)in_sizes; (void)n_in; (void)out_size; (void)ws_size;

    const float* self = (const float*)d_in[0];
    const float* c0   = (const float*)d_in[1];
    const float* c1   = (const float*)d_in[2];
    const float* Wq   = (const float*)d_in[3];  const float* bq   = (const float*)d_in[4];
    const float* Wk   = (const float*)d_in[5];  const float* bk   = (const float*)d_in[6];
    const float* Wv   = (const float*)d_in[7];  const float* bv   = (const float*)d_in[8];
    const float* Wck0 = (const float*)d_in[9];  const float* bck0 = (const float*)d_in[10];
    const float* Wcv0 = (const float*)d_in[11]; const float* bcv0 = (const float*)d_in[12];
    const float* Wck1 = (const float*)d_in[13]; const float* bck1 = (const float*)d_in[14];
    const float* Wcv1 = (const float*)d_in[15]; const float* bcv1 = (const float*)d_in[16];
    const float* Wo   = (const float*)d_in[17]; const float* bo   = (const float*)d_in[18];
    const float* ctxb = (const float*)d_in[19];
    const unsigned char* mS = (const unsigned char*)d_in[20];
    const unsigned char* m0 = (const unsigned char*)d_in[21];
    const unsigned char* m1 = (const unsigned char*)d_in[22];

    float* outP  = (float*)d_out;                               // (B,S,D)     4,194,304
    float* topP  = outP + (size_t)Bb * Ss * Dn;                 // (B,S,1024)  4,194,304
    float* attnP = topP + (size_t)Bb * Ss * (C1n + C2n);        // (B,H,S,KT)  134,217,728

    size_t off = 0;
    auto wsp = [&](size_t bytes) {
        void* p = (char*)d_ws + off;
        off += (bytes + 255) & ~(size_t)255;
        return p;
    };
    ushort_t* xs  = (ushort_t*)wsp((size_t)Bb * Ss * Dn * 2);
    ushort_t* x0  = (ushort_t*)wsp((size_t)Bb * C1n * Dn * 2);
    ushort_t* x1  = (ushort_t*)wsp((size_t)Bb * C2n * Dn * 2);
    ushort_t* WqT = (ushort_t*)wsp((size_t)Dn * Dn * 2);
    ushort_t* WkT = (ushort_t*)wsp((size_t)Dn * Dn * 2);
    ushort_t* WvT = (ushort_t*)wsp((size_t)Dn * Dn * 2);
    ushort_t* W0kT= (ushort_t*)wsp((size_t)Dn * Dn * 2);
    ushort_t* W0vT= (ushort_t*)wsp((size_t)Dn * Dn * 2);
    ushort_t* W1kT= (ushort_t*)wsp((size_t)Dn * Dn * 2);
    ushort_t* W1vT= (ushort_t*)wsp((size_t)Dn * Dn * 2);
    ushort_t* WoT = (ushort_t*)wsp((size_t)Dn * Dn * 2);
    ushort_t* qb  = (ushort_t*)wsp((size_t)Bb * Hn * Ss  * DHn * 2);
    ushort_t* kb  = (ushort_t*)wsp((size_t)Bb * Hn * KTn * DHn * 2);
    ushort_t* vt  = (ushort_t*)wsp((size_t)Bb * Hn * DHn * KTn * 2);
    ushort_t* cx  = (ushort_t*)wsp((size_t)Bb * Ss * Dn * 2);
    float*    st  = (float*)   wsp((size_t)Bb * Hn * Ss * 4 * 4);

    // 1. convert weights (transposed) and activations to bf16
    cvt_tr_kernel<<<4096, 256, 0, stream>>>(Wq,   WqT);
    cvt_tr_kernel<<<4096, 256, 0, stream>>>(Wk,   WkT);
    cvt_tr_kernel<<<4096, 256, 0, stream>>>(Wv,   WvT);
    cvt_tr_kernel<<<4096, 256, 0, stream>>>(Wck0, W0kT);
    cvt_tr_kernel<<<4096, 256, 0, stream>>>(Wcv0, W0vT);
    cvt_tr_kernel<<<4096, 256, 0, stream>>>(Wck1, W1kT);
    cvt_tr_kernel<<<4096, 256, 0, stream>>>(Wcv1, W1vT);
    cvt_tr_kernel<<<4096, 256, 0, stream>>>(Wo,   WoT);
    cvt_bf16_kernel<<<16384, 256, 0, stream>>>(self, xs, Bb * Ss * Dn);
    cvt_bf16_kernel<<<8192, 256, 0, stream>>>(c0, x0, Bb * C1n * Dn);
    cvt_bf16_kernel<<<8192, 256, 0, stream>>>(c1, x1, Bb * C2n * Dn);

    // 2. projections (WMMA GEMMs, 16x64 tile per wave)
    dim3 blk(128, 1, 1);
    dim3 g4096(4, 256, 1), g2048(4, 128, 1);
    gemm16_kernel<<<g4096, blk, 0, stream>>>(xs, WqT,  bq,   Ss,  0,          1, SCALE_Q, qb, nullptr);
    gemm16_kernel<<<g4096, blk, 0, stream>>>(xs, WkT,  bk,   Ss,  0,          2, 1.f, kb, nullptr);
    gemm16_kernel<<<g2048, blk, 0, stream>>>(x0, W0kT, bck0, C1n, Ss,         2, 1.f, kb, nullptr);
    gemm16_kernel<<<g2048, blk, 0, stream>>>(x1, W1kT, bck1, C2n, Ss + C1n,   2, 1.f, kb, nullptr);
    gemm16_kernel<<<g4096, blk, 0, stream>>>(xs, WvT,  bv,   Ss,  0,          3, 1.f, vt, nullptr);
    gemm16_kernel<<<g2048, blk, 0, stream>>>(x0, W0vT, bcv0, C1n, Ss,         3, 1.f, vt, nullptr);
    gemm16_kernel<<<g2048, blk, 0, stream>>>(x1, W1vT, bcv1, C2n, Ss + C1n,   3, 1.f, vt, nullptr);

    // 3. flash attention (raw scores into attn region + stats + ctx)
    attn_kernel<<<1024, 128, 0, stream>>>(qb, kb, vt, mS, m0, m1, ctxb, attnP, st, cx);

    // 4. in-place softmax normalization + top_attn (head 0)
    norm_attn_kernel<<<524288, 256, 0, stream>>>(attnP, st, topP);

    // 5. output projection -> d_out
    gemm16_kernel<<<g4096, blk, 0, stream>>>(cx, WoT, bo, Ss, 0, 0, 1.f, nullptr, outP);
}
